// AwesomeLSTM_9216999817258
// MI455X (gfx1250) — matrix operand order repaired and checked by reading, not compiled, not run on hardware
//
#include <hip/hip_runtime.h>
#include <hip/hip_bf16.h>
#include <math.h>

// ---------------------------------------------------------------------------
// Sizes (fixed by the reference problem)
// ---------------------------------------------------------------------------
#define SEQ   256
#define BATCH 64
#define DIM   1024
#define UNITS 1024
#define GCOLS (4 * UNITS)          // 4096 gate columns
#define SB    (SEQ * BATCH)        // 16384 rows of x
#define NBLK  (UNITS / 16)         // 64 persistent blocks, one per WGP
#define LPAD  8                    // +8 bf16 row pad -> 2064B stride, bank-conflict-free
#define LSTRIDE (1024 + LPAD)

typedef __attribute__((ext_vector_type(16))) __bf16 v16bf;
typedef __attribute__((ext_vector_type(8)))  float  v8f;

union FragU {
    v16bf v;
    uint4 q[2];
};

// ---------------------------------------------------------------------------
// f32 -> bf16 conversion (grid-stride)
// ---------------------------------------------------------------------------
__global__ __launch_bounds__(256) void f32_to_bf16_kernel(
    const float* __restrict__ src, __bf16* __restrict__ dst, int n) {
    int i = blockIdx.x * blockDim.x + threadIdx.x;
    int stride = gridDim.x * blockDim.x;
    for (; i < n; i += stride) dst[i] = (__bf16)src[i];
}

// Initialize recurrent state + zero the grid-barrier counter
__global__ __launch_bounds__(256) void init_state_kernel(
    const float* __restrict__ hx, const float* __restrict__ cx,
    __bf16* __restrict__ h0, float* __restrict__ c, int n,
    unsigned* __restrict__ counter) {
    if (blockIdx.x == 0 && threadIdx.x == 0) *counter = 0u;
    int i = blockIdx.x * blockDim.x + threadIdx.x;
    int stride = gridDim.x * blockDim.x;
    for (; i < n; i += stride) {
        h0[i] = (__bf16)hx[i];
        c[i]  = cx[i];
    }
}

// ---------------------------------------------------------------------------
// Phase 1: x_proj = x @ W_ih^T + (b_ih + b_hh)
//   64x64 macro-tile per wave: 4x4 grid of 16x16 WMMA tiles, 4x fragment reuse.
//   A = x_bf16 [SB x DIM] row major; B = W_ih row n = weights over k (contig).
// ---------------------------------------------------------------------------
__global__ __launch_bounds__(256) void xproj_wmma_kernel(
    const __bf16* __restrict__ xbf, const __bf16* __restrict__ wih,
    const float* __restrict__ b_ih, const float* __restrict__ b_hh,
    float* __restrict__ xp) {
    const int lane = threadIdx.x & 31;
    const int wid  = threadIdx.x >> 5;
    const int gw   = blockIdx.x * 8 + wid;   // global wave id
    const int NT   = GCOLS / 64;             // 64 macro-cols
    const int mt   = gw / NT;                // 0 .. SB/64-1 (256)
    const int nt   = gw % NT;                // 0 .. 63
    const int l16  = lane & 15;
    const int half = lane >> 4;

    const __bf16* a0 = xbf + (size_t)(mt * 64 + l16) * DIM;  // A row = lane%16 (+sm*16)
    const __bf16* b0 = wih + (size_t)(nt * 64 + l16) * DIM;  // B col = lane%16 (+sn*16)

    v8f acc[4][4] = {};
    for (int kb = 0; kb < DIM; kb += 32) {
        FragU a[4], b[4];
#pragma unroll
        for (int s = 0; s < 4; ++s) {
            const __bf16* ap = a0 + (size_t)s * 16 * DIM + kb;
            a[s].q[0] = *(const uint4*)(ap + half * 8);
            a[s].q[1] = *(const uint4*)(ap + 16 + half * 8);
            const __bf16* bp = b0 + (size_t)s * 16 * DIM + kb;
            b[s].q[0] = *(const uint4*)(bp + half * 8);
            b[s].q[1] = *(const uint4*)(bp + 16 + half * 8);
        }
#pragma unroll
        for (int sm = 0; sm < 4; ++sm)
#pragma unroll
            for (int sn = 0; sn < 4; ++sn)
                acc[sm][sn] = __builtin_amdgcn_wmma_f32_16x16x32_bf16(
                    false, a[sm].v, false, b[sn].v, (short)0, acc[sm][sn],
                    false, false);
    }

#pragma unroll
    for (int sn = 0; sn < 4; ++sn) {
        const int ncol = nt * 64 + sn * 16 + l16;
        const float bias = b_ih[ncol] + b_hh[ncol];
#pragma unroll
        for (int sm = 0; sm < 4; ++sm) {
            const int row0 = mt * 64 + sm * 16 + half * 8;  // D row = r + half*8
#pragma unroll
            for (int r = 0; r < 8; ++r)
                xp[(size_t)(row0 + r) * GCOLS + ncol] = acc[sm][sn][r] + bias;
        }
    }
}

// ---------------------------------------------------------------------------
// Phase 2: persistent scan kernel. 64 blocks (one per WGP), 512 threads each.
//   Block owns units [u0,u0+16). Its W_hh slice (4 gates x 16 rows x 1024 k,
//   129 KB padded) lives in LDS for ALL 256 timesteps. Per step, h (129 KB,
//   mask folded in) is staged to LDS, 16 waves run the 4-gate GEMM from LDS,
//   gates go through LDS to the fused sigmoid/tanh + c/h update. Timesteps
//   are separated by a monotonic atomic grid barrier; h is double-buffered
//   in global memory (only cross-block state).
// ---------------------------------------------------------------------------
__global__ __launch_bounds__(512) void lstm_scan_kernel(
    const float*  __restrict__ xp,        // [SB, GCOLS]
    const __bf16* __restrict__ whh,       // [GCOLS, UNITS]
    __bf16* __restrict__ hbuf0,           // [BATCH, UNITS] double buffer
    __bf16* __restrict__ hbuf1,
    float*  __restrict__ c_state,         // [BATCH, UNITS]
    const float* __restrict__ reset,      // [SEQ*BATCH]
    float* __restrict__ hs0, float* __restrict__ hs1, float* __restrict__ cs,
    unsigned* __restrict__ counter) {
    __shared__ __bf16 w_sh[64][LSTRIDE];  // 129 KB: rows = g*16 + unit_local
    __shared__ __bf16 h_sh[64][LSTRIDE];  // 129 KB: rows = batch
    __shared__ float  gsh[4][BATCH][16];  // 16 KB gate staging

    const int tid  = threadIdx.x;
    const int lane = tid & 31;
    const int wid  = tid >> 5;            // 0..15
    const int g    = wid >> 2;            // gate 0..3 (i,f,g,o)
    const int mt   = wid & 3;             // M-tile (16 batch rows)
    const int u0   = blockIdx.x * 16;
    const int l16  = lane & 15;
    const int half = lane >> 4;

    // ---- stage this block's W_hh slice into LDS once (64 rows x 2KB) ----
    for (int idx = tid; idx < 64 * 128; idx += 512) {
        const int r  = idx >> 7;                       // local row 0..63
        const int ch = (idx & 127) * 8;                // 8-bf16 chunk
        const int grow = (r >> 4) * UNITS + u0 + (r & 15);
        *(uint4*)(&w_sh[r][ch]) = *(const uint4*)(whh + (size_t)grow * UNITS + ch);
    }
    __syncthreads();

    const __bf16* ap = &h_sh[mt * 16 + l16][0];        // A: batch row per lane
    const __bf16* bp = &w_sh[g * 16 + l16][0];         // B: W row per lane

    for (int t = 0; t < SEQ; ++t) {
        const __bf16* hin  = (t & 1) ? hbuf1 : hbuf0;
        __bf16*       hout = (t & 1) ? hbuf0 : hbuf1;

        // ---- stage h (episode mask folded in: row zeroed on reset) ----
        for (int idx = tid; idx < 64 * 128; idx += 512) {
            const int r  = idx >> 7;                   // batch row
            const int ch = (idx & 127) * 8;
            uint4 v = *(const uint4*)(hin + (size_t)r * UNITS + ch);
            if (reset[t * BATCH + r] != 0.0f) v = make_uint4(0u, 0u, 0u, 0u);
            *(uint4*)(&h_sh[r][ch]) = v;
        }
        __syncthreads();

        // ---- 16x16 WMMA tile per wave, K=1024 from LDS ----
        v8f acc = {};
        for (int kb = 0; kb < UNITS; kb += 32) {
            FragU a, b;
            a.q[0] = *(const uint4*)(ap + kb + half * 8);
            a.q[1] = *(const uint4*)(ap + kb + 16 + half * 8);
            b.q[0] = *(const uint4*)(bp + kb + half * 8);
            b.q[1] = *(const uint4*)(bp + kb + 16 + half * 8);
            acc = __builtin_amdgcn_wmma_f32_16x16x32_bf16(
                false, a.v, false, b.v, (short)0, acc, false, false);
        }

        // ---- add x-projection (includes biases), stage gates ----
        const int rowbase = mt * 16 + half * 8;
        const int gcol = g * UNITS + u0 + l16;
#pragma unroll
        for (int r = 0; r < 8; ++r) {
            const int row = rowbase + r;
            gsh[g][row][l16] = acc[r] + xp[(size_t)(t * BATCH + row) * GCOLS + gcol];
        }
        __syncthreads();

        // ---- fused elementwise: 64x16 cells, 512 threads -> 2 each ----
        for (int idx = tid; idx < BATCH * 16; idx += 512) {
            const int b  = idx >> 4;
            const int ul = idx & 15;
            const float i_ = 1.0f / (1.0f + __expf(-gsh[0][b][ul]));
            const float f_ = 1.0f / (1.0f + __expf(-gsh[1][b][ul]));
            const float g_ = tanhf(gsh[2][b][ul]);
            const float o_ = 1.0f / (1.0f + __expf(-gsh[3][b][ul]));
            const float m  = 1.0f - reset[t * BATCH + b];
            const size_t su = (size_t)b * UNITS + u0 + ul;
            const float cn = f_ * (c_state[su] * m) + i_ * g_;
            const float hn = o_ * tanhf(cn);
            c_state[su] = cn;
            hout[su]    = (__bf16)hn;
            const size_t off = (size_t)(t * BATCH + b) * UNITS + u0 + ul;
            hs0[off] = hn;
            hs1[off] = hn;
            cs[off]  = cn;
        }

        // ---- grid-wide barrier between timesteps ----
        __syncthreads();
        if (tid == 0) {
            __threadfence();                           // release h/c writes
            atomicAdd(counter, 1u);
            const unsigned target = (unsigned)(t + 1) * NBLK;
            while (__hip_atomic_load(counter, __ATOMIC_ACQUIRE,
                                     __HIP_MEMORY_SCOPE_AGENT) < target)
                __builtin_amdgcn_s_sleep(1);
        }
        __syncthreads();
        __threadfence();                               // invalidate stale lines
    }
}

// ---------------------------------------------------------------------------
// Host-side orchestration
// ---------------------------------------------------------------------------
extern "C" void kernel_launch(void* const* d_in, const int* in_sizes, int n_in,
                              void* d_out, int out_size, void* d_ws, size_t ws_size,
                              hipStream_t stream) {
    const float* x     = (const float*)d_in[0];   // [SEQ,BATCH,DIM]
    const float* hx    = (const float*)d_in[1];   // [BATCH,UNITS]
    const float* cx    = (const float*)d_in[2];   // [BATCH,UNITS]
    const float* reset = (const float*)d_in[3];   // [SEQ,BATCH,1]
    const float* W_ih  = (const float*)d_in[4];   // [GCOLS,DIM]
    const float* W_hh  = (const float*)d_in[5];   // [GCOLS,UNITS]
    const float* b_ih  = (const float*)d_in[6];   // [GCOLS]
    const float* b_hh  = (const float*)d_in[7];   // [GCOLS]

    // Workspace layout (regions naturally 256B-aligned)
    char* ws = (char*)d_ws;
    float*    xp     = (float*)(ws);                                    // 256 MB
    __bf16*   x_bf   = (__bf16*)(ws + (size_t)SB * GCOLS * 4);          //  32 MB
    __bf16*   wih_bf = (__bf16*)((char*)x_bf + (size_t)SB * DIM * 2);   //   8 MB
    __bf16*   whh_bf = (__bf16*)((char*)wih_bf + (size_t)GCOLS * DIM * 2);
    __bf16*   h_buf0 = (__bf16*)((char*)whh_bf + (size_t)GCOLS * UNITS * 2);
    __bf16*   h_buf1 = (__bf16*)((char*)h_buf0 + (size_t)BATCH * UNITS * 2);
    float*    cstate = (float*)((char*)h_buf1 + (size_t)BATCH * UNITS * 2);
    unsigned* counter = (unsigned*)((char*)cstate + (size_t)BATCH * UNITS * 4);

    // bf16 conversions (bandwidth-trivial vs. total traffic)
    f32_to_bf16_kernel<<<4096, 256, 0, stream>>>(x, x_bf, SB * DIM);
    f32_to_bf16_kernel<<<2048, 256, 0, stream>>>(W_ih, wih_bf, GCOLS * DIM);
    f32_to_bf16_kernel<<<2048, 256, 0, stream>>>(W_hh, whh_bf, GCOLS * UNITS);
    init_state_kernel<<<256, 256, 0, stream>>>(hx, cx, h_buf0, cstate,
                                               BATCH * UNITS, counter);

    // Phase 1: register-blocked input-projection GEMM (64x64 tile per wave)
    {
        const int waves = (SB / 64) * (GCOLS / 64);   // 16384 macro-tiles
        xproj_wmma_kernel<<<waves / 8, 256, 0, stream>>>(x_bf, wih_bf, b_ih, b_hh, xp);
    }

    // Phase 2: one persistent kernel for the whole 256-step scan
    float* hs0 = (float*)d_out;
    float* hs1 = hs0 + (size_t)SEQ * BATCH * UNITS;
    float* cs  = hs1 + (size_t)SEQ * BATCH * UNITS;
    lstm_scan_kernel<<<NBLK, 512, 0, stream>>>(
        xp, whh_bf, h_buf0, h_buf1, cstate, reset, hs0, hs1, cs, counter);
}
